// NonLocalBlock1D_41540923687169
// MI455X (gfx1250) — matrix operand order repaired and checked by reading, not compiled, not run on hardware
//
#include <hip/hip_runtime.h>
#include <hip/hip_bf16.h>

typedef __attribute__((ext_vector_type(16))) _Float16 v16h;
typedef __attribute__((ext_vector_type(8)))  _Float16 v8h;
typedef __attribute__((ext_vector_type(8)))  float    v8f;

#define C_DIM 512
#define CI_DIM 256
#define NTOK 4096
#define NB 8
#define BN (NB * NTOK)   // 32768 tokens total
#define KB 32            // keys per flash step

#define WMMA_F16(A, B, Cacc) \
  __builtin_amdgcn_wmma_f32_16x16x32_f16(false, (A), false, (B), (short)0, (Cacc), false, false)

// A-fragment (16x32 f16, MxK): lane<16 -> row=lane, K elems k..k+15;
// lane>=16 -> row=lane-16, K elems k+16..k+31.
// Caller passes p already offset by (lane<16?0:16); we grab 8 f16 at p and 8 at p+8.
static __device__ __forceinline__ v16h ldA16(const _Float16* p) {
  v8h a = *(const v8h*)(p);
  v8h b = *(const v8h*)(p + 8);
  v16h r;
#pragma unroll
  for (int i = 0; i < 8; ++i) { r[i] = a[i]; r[i + 8] = b[i]; }
  return r;
}

// B-fragment (32x16 f16, KxN): lane<16 -> col=lane, K=0..15; lane>=16 -> col=lane-16, K=16..31.
// Caller passes p pointing at 16 contiguous f16 for this lane.
static __device__ __forceinline__ v16h ldB16(const _Float16* p) {
  v8h a = *(const v8h*)(p);
  v8h b = *(const v8h*)(p + 8);
  v16h r;
#pragma unroll
  for (int i = 0; i < 8; ++i) { r[i] = a[i]; r[i + 8] = b[i]; }
  return r;
}

// Async global -> LDS copy, 16 bytes per lane. ASYNCcnt-tracked (fire and forget).
// lds_off = wave-relative LDS byte offset (low 32 bits of the generic pointer).
static __device__ __forceinline__ void async_cp16(_Float16* lds_dst, const _Float16* gsrc) {
  unsigned int lds_off = (unsigned int)(unsigned long long)lds_dst;
  asm volatile("global_load_async_to_lds_b128 %0, %1, off"
               :: "v"(lds_off), "v"(gsrc)
               : "memory");
}

static __device__ __forceinline__ void wait_async0() {
#if __has_builtin(__builtin_amdgcn_s_wait_asynccnt)
  __builtin_amdgcn_s_wait_asynccnt(0);
#else
  asm volatile("s_wait_asynccnt 0" ::: "memory");
#endif
}

// ---------------- Stage 0: dtype conversion / weight transposes ----------------

__global__ void k_cvt_f16(const float* __restrict__ src, _Float16* __restrict__ dst, int n) {
  int i = blockIdx.x * blockDim.x + threadIdx.x;
  if (i < n) dst[i] = (_Float16)src[i];
}

// W: rows x cols (row-major f32) -> WT: cols x rows (row-major f16)
__global__ void k_transpose_f16(const float* __restrict__ W, _Float16* __restrict__ WT,
                                int rows, int cols) {
  int i = blockIdx.x * blockDim.x + threadIdx.x;
  if (i >= rows * cols) return;
  int n = i / rows;
  int k = i - n * rows;
  WT[(size_t)n * rows + k] = (_Float16)W[(size_t)k * cols + n];
}

// ---------------- Stage 1: fused g/theta/phi projections ----------------
// grid.x = BN/16 (16-token tiles), grid.y = 3 (which projection)
// block = 256 threads = 8 waves; wave w covers output cols [w*32, w*32+32)
__global__ __launch_bounds__(256) void k_proj(
    const _Float16* __restrict__ xh,
    const _Float16* __restrict__ WgT, const _Float16* __restrict__ WtT,
    const _Float16* __restrict__ WpT,
    const float* __restrict__ bg, const float* __restrict__ bt, const float* __restrict__ bp,
    _Float16* __restrict__ gT, _Float16* __restrict__ th, _Float16* __restrict__ ph) {
  const int lane = threadIdx.x & 31;
  const int wave = threadIdx.x >> 5;
  const int m0 = blockIdx.x * 16;
  const int which = blockIdx.y;
  const _Float16* WT = (which == 0) ? WgT : ((which == 1) ? WtT : WpT);
  const float* bias   = (which == 0) ? bg  : ((which == 1) ? bt  : bp);
  const int n0 = wave * 32;
  const int halfsel = (lane < 16) ? 0 : 1;
  const int lrow = lane & 15;

  v8f acc0 = {}, acc1 = {};
  const _Float16* arow = xh + (size_t)(m0 + lrow) * C_DIM + halfsel * 16;
  const _Float16* b0p  = WT + (size_t)(n0 + lrow) * C_DIM + halfsel * 16;
  const _Float16* b1p  = b0p + (size_t)16 * C_DIM;

#pragma unroll 4
  for (int k = 0; k < C_DIM; k += 32) {
    v16h a  = ldA16(arow + k);
    v16h f0 = ldB16(b0p + k);
    v16h f1 = ldB16(b1p + k);
    acc0 = WMMA_F16(a, f0, acc0);
    acc1 = WMMA_F16(a, f1, acc1);
  }

#pragma unroll
  for (int v = 0; v < 8; ++v) {
    int row = v + halfsel * 8;
    int m = m0 + row;
    int bb = m >> 12;
    int t  = m & (NTOK - 1);
    int nA = n0 + lrow, nB = n0 + 16 + lrow;
    float eA = acc0[v] + bias[nA];
    float eB = acc1[v] + bias[nB];
    if (which == 0) {
      // g stored channel-major (transposed) so attention P@G B-fragments are contiguous
      gT[((size_t)bb * CI_DIM + nA) * NTOK + t] = (_Float16)eA;
      gT[((size_t)bb * CI_DIM + nB) * NTOK + t] = (_Float16)eB;
    } else {
      _Float16* dst = (which == 1) ? th : ph;
      dst[(size_t)m * CI_DIM + nA] = (_Float16)eA;
      dst[(size_t)m * CI_DIM + nB] = (_Float16)eB;
    }
  }
}

// ---------------- Stage 2: flash attention ----------------
// grid.x = NTOK/64 (4 query tiles per block), grid.y = batch; block = 128 = 4 waves.
// Key (phi) and value (gT) blocks are staged cooperatively into LDS with
// double-buffered async global->LDS copies; all 4 waves consume them from LDS.
__global__ __launch_bounds__(128) void k_flash(
    const _Float16* __restrict__ th, const _Float16* __restrict__ ph,
    const _Float16* __restrict__ gT, _Float16* __restrict__ yh) {
  __shared__ __align__(16) _Float16 kbuf[2][KB * CI_DIM];   // keys: [key][channel], 2 x 16KB
  __shared__ __align__(16) _Float16 gbuf[2][CI_DIM * KB];   // values: [channel][key], 2 x 16KB
  __shared__ __align__(16) _Float16 pbuf[4][16 * KB];       // per-wave P staging, 4KB
  const int tid  = threadIdx.x;
  const int lane = threadIdx.x & 31;
  const int wave = threadIdx.x >> 5;
  const int bb = blockIdx.y;
  const int q0 = (blockIdx.x * 4 + wave) * 16;
  const int halfsel = (lane < 16) ? 0 : 1;
  const int lrow = lane & 15;

  // Cooperative stage of one 32-key block (32KB): 16 x 16B async copies per thread.
  auto stage = [&](int buf, int kb) {
#pragma unroll
    for (int c = 0; c < 8; ++c) {
      int chunk = tid + c * 128;           // 0..1023, 8 f16 each
      int r   = chunk >> 5;                // key row 0..31
      int col = (chunk & 31) * 8;          // channel 0..255
      async_cp16(&kbuf[buf][r * CI_DIM + col],
                 ph + ((size_t)bb * NTOK + kb + r) * CI_DIM + col);
    }
#pragma unroll
    for (int c = 0; c < 8; ++c) {
      int chunk = tid + c * 128;
      int ch = chunk >> 2;                 // channel 0..255
      int k  = (chunk & 3) * 8;            // key 0..31
      async_cp16(&gbuf[buf][ch * KB + k],
                 gT + ((size_t)bb * CI_DIM + ch) * NTOK + kb + k);
    }
  };

  // Preload Q fragments (16x256 = 8 chunks of K=32) — stays in VGPRs for all key blocks
  v16h qf[8];
  const _Float16* qrow = th + ((size_t)bb * NTOK + q0 + lrow) * CI_DIM + halfsel * 16;
#pragma unroll
  for (int i = 0; i < 8; ++i) qf[i] = ldA16(qrow + i * 32);

  v8f zero = {};
  v8f yacc[16];
#pragma unroll
  for (int t = 0; t < 16; ++t) yacc[t] = zero;
  float m_r[8], l_r[8];
#pragma unroll
  for (int v = 0; v < 8; ++v) { m_r[v] = -3.0e38f; l_r[v] = 0.0f; }

  _Float16* pb = pbuf[wave];

  stage(0, 0);
  wait_async0();
  __syncthreads();

  for (int kb = 0; kb < NTOK; kb += KB) {
    const int cur = (kb >> 5) & 1;
    if (kb + KB < NTOK) stage(cur ^ 1, kb + KB);   // prefetch next block (async)

    // S = Q @ K^T : two 16x16 key tiles, K-dim = 256 channels in 8 chunks (from LDS)
    v8f s0 = zero, s1 = zero;
    const _Float16* k0p = &kbuf[cur][lrow * CI_DIM + halfsel * 16];
    const _Float16* k1p = k0p + 16 * CI_DIM;
#pragma unroll
    for (int i = 0; i < 8; ++i) {
      v16h b0 = ldB16(k0p + i * 32);
      v16h b1 = ldB16(k1p + i * 32);
      s0 = WMMA_F16(qf[i], b0, s0);
      s1 = WMMA_F16(qf[i], b1, s1);
    }

    // Online softmax. C layout: vgpr v, lane<16 -> row=v, lane>=16 -> row=v+8;
    // a full S row lives in one 16-lane half, so half-wave reductions suffice.
#pragma unroll
    for (int v = 0; v < 8; ++v) {
      float cm = fmaxf(s0[v], s1[v]);
#pragma unroll
      for (int off = 8; off >= 1; off >>= 1) cm = fmaxf(cm, __shfl_xor(cm, off, 32));
      float mnew = fmaxf(m_r[v], cm);
      float sc = __expf(m_r[v] - mnew);
      float p0 = __expf(s0[v] - mnew);
      float p1 = __expf(s1[v] - mnew);
      float rs = p0 + p1;
#pragma unroll
      for (int off = 8; off >= 1; off >>= 1) rs += __shfl_xor(rs, off, 32);
      l_r[v] = l_r[v] * sc + rs;
      m_r[v] = mnew;
#pragma unroll
      for (int t = 0; t < 16; ++t) yacc[t][v] *= sc;
      // stage P (f16) to LDS so we can re-read it in A-fragment layout
      int prow = v + halfsel * 8;
      pb[prow * KB + lrow]      = (_Float16)p0;
      pb[prow * KB + 16 + lrow] = (_Float16)p1;
    }

    // P fragment (16 queries x 32 keys), then y += P @ G (16 channel tiles, reuse P)
    v16h pf = ldA16(pb + lrow * KB + halfsel * 16);
#pragma unroll
    for (int t = 0; t < 16; ++t) {
      v16h gf = ldB16(&gbuf[cur][(t * 16 + lrow) * KB + halfsel * 16]);
      yacc[t] = WMMA_F16(pf, gf, yacc[t]);
    }

    wait_async0();      // next block fully staged
    __syncthreads();    // and everyone is done reading the current one
  }

  // finalize: y /= row-sum, store f16 row-major
#pragma unroll
  for (int v = 0; v < 8; ++v) {
    float inv = 1.0f / l_r[v];
    int m = q0 + v + halfsel * 8;
#pragma unroll
    for (int t = 0; t < 16; ++t) {
      yh[((size_t)bb * NTOK + m) * CI_DIM + t * 16 + lrow] = (_Float16)(yacc[t][v] * inv);
    }
  }
}

// ---------------- Stage 3: output projection + bias + residual ----------------
// grid = BN/16, block = 256 = 8 waves; wave w covers cols [w*64, w*64+64)
__global__ __launch_bounds__(256) void k_outproj(
    const _Float16* __restrict__ yh, const _Float16* __restrict__ WoT,
    const float* __restrict__ bo, const float* __restrict__ x,
    float* __restrict__ out) {
  const int lane = threadIdx.x & 31;
  const int wave = threadIdx.x >> 5;
  const int m0 = blockIdx.x * 16;
  const int n0 = wave * 64;
  const int halfsel = (lane < 16) ? 0 : 1;
  const int lrow = lane & 15;

  v8f zero = {};
  v8f acc[4];
#pragma unroll
  for (int a4 = 0; a4 < 4; ++a4) acc[a4] = zero;

  const _Float16* arow = yh + (size_t)(m0 + lrow) * CI_DIM + halfsel * 16;
#pragma unroll 2
  for (int k = 0; k < CI_DIM; k += 32) {
    v16h a = ldA16(arow + k);
#pragma unroll
    for (int a4 = 0; a4 < 4; ++a4) {
      const _Float16* bp =
          WoT + (size_t)(n0 + a4 * 16 + lrow) * CI_DIM + halfsel * 16 + k;
      v16h bf = ldB16(bp);
      acc[a4] = WMMA_F16(a, bf, acc[a4]);
    }
  }

#pragma unroll
  for (int a4 = 0; a4 < 4; ++a4) {
    int n = n0 + a4 * 16 + lrow;
    float bias = bo[n];
#pragma unroll
    for (int v = 0; v < 8; ++v) {
      int m = m0 + v + halfsel * 8;
      size_t idx = (size_t)m * C_DIM + n;
      out[idx] = acc[a4][v] + bias + x[idx];
    }
  }
}

// ---------------- launcher ----------------

extern "C" void kernel_launch(void* const* d_in, const int* in_sizes, int n_in,
                              void* d_out, int out_size, void* d_ws, size_t ws_size,
                              hipStream_t stream) {
  const float* x  = (const float*)d_in[0];
  const float* Wg = (const float*)d_in[1];
  const float* bg = (const float*)d_in[2];
  const float* Wt = (const float*)d_in[3];
  const float* bt = (const float*)d_in[4];
  const float* Wp = (const float*)d_in[5];
  const float* bp = (const float*)d_in[6];
  const float* Wo = (const float*)d_in[7];
  const float* bo = (const float*)d_in[8];
  float* out = (float*)d_out;

  char* w = (char*)d_ws;
  _Float16* xh  = (_Float16*)w; w += (size_t)BN * C_DIM * sizeof(_Float16);
  _Float16* WgT = (_Float16*)w; w += (size_t)CI_DIM * C_DIM * sizeof(_Float16);
  _Float16* WtT = (_Float16*)w; w += (size_t)CI_DIM * C_DIM * sizeof(_Float16);
  _Float16* WpT = (_Float16*)w; w += (size_t)CI_DIM * C_DIM * sizeof(_Float16);
  _Float16* WoT = (_Float16*)w; w += (size_t)C_DIM * CI_DIM * sizeof(_Float16);
  _Float16* gT  = (_Float16*)w; w += (size_t)BN * CI_DIM * sizeof(_Float16);
  _Float16* th  = (_Float16*)w; w += (size_t)BN * CI_DIM * sizeof(_Float16);
  _Float16* ph  = (_Float16*)w; w += (size_t)BN * CI_DIM * sizeof(_Float16);
  _Float16* yh  = (_Float16*)w; w += (size_t)BN * CI_DIM * sizeof(_Float16);

  // Stage 0: conversions
  {
    int n = BN * C_DIM;
    k_cvt_f16<<<(n + 255) / 256, 256, 0, stream>>>(x, xh, n);
    int nw = C_DIM * CI_DIM;
    k_transpose_f16<<<(nw + 255) / 256, 256, 0, stream>>>(Wg, WgT, C_DIM, CI_DIM);
    k_transpose_f16<<<(nw + 255) / 256, 256, 0, stream>>>(Wt, WtT, C_DIM, CI_DIM);
    k_transpose_f16<<<(nw + 255) / 256, 256, 0, stream>>>(Wp, WpT, C_DIM, CI_DIM);
    k_transpose_f16<<<(nw + 255) / 256, 256, 0, stream>>>(Wo, WoT, CI_DIM, C_DIM);
  }
  // Stage 1: projections (g transposed, theta/phi row-major)
  k_proj<<<dim3(BN / 16, 3), 256, 0, stream>>>(xh, WgT, WtT, WpT, bg, bt, bp, gT, th, ph);
  // Stage 2: flash attention (async double-buffered LDS staging)
  k_flash<<<dim3(NTOK / 64, NB), 128, 0, stream>>>(th, ph, gT, yh);
  // Stage 3: output projection + residual
  k_outproj<<<BN / 16, 256, 0, stream>>>(yh, WoT, bo, x, out);
}
